// GVP_MPNN_7937099563208
// MI455X (gfx1250) — matrix operand order repaired and checked by reading, not compiled, not run on hardware
//
#include <hip/hip_runtime.h>
#include <hip/hip_bf16.h>
#include <math.h>

typedef __attribute__((ext_vector_type(16))) _Float16 v16h;
typedef __attribute__((ext_vector_type(8)))  float    v8f;

#define N_NODES 50000
#define N_EDGES 800000
#define NODE_DIM 96    // 48 scalars + 16*3 vector
#define EDGE_DIM 80    // 32 scalars + 16*3 vector
#define WPB 4          // waves (wave32) per block

__device__ __forceinline__ float fsqrt(float x)  { return __builtin_amdgcn_sqrtf(x); }
__device__ __forceinline__ float frsqrt(float x) { return __builtin_amdgcn_rsqf(x); }

__device__ __forceinline__ v8f wmma_f16(v16h a, v16h b, v8f c) {
  return __builtin_amdgcn_wmma_f32_16x16x32_f16(false, a, false, b, (short)0, c, false, false);
}

__device__ __forceinline__ float f4_elem(float4 v, int j) {
  return (j == 0) ? v.x : (j == 1) ? v.y : (j == 2) ? v.z : v.w;
}

// A operand (16x32 f16, M on lanes): lane<16: K=k0+0..15 ; lane>=16: K=k0+16..31
__device__ __forceinline__ v16h lds_tileA(const __attribute__((address_space(3))) _Float16* p0,
                                          int kpad, int kbase) {
  int lane = threadIdx.x & 31;
  const __attribute__((address_space(3))) _Float16* p =
      p0 + (lane & 15) * kpad + kbase + ((lane & 16) ? 16 : 0);
  v16h a;
#pragma unroll
  for (int i = 0; i < 16; ++i) a[i] = p[i];
  return a;
}

// B operand (32x16 f16 = W^T, N on lanes): lane<16: K=kbase+0..15 ; lane>=16: K=kbase+16..31
__device__ __forceinline__ v16h lds_tileB(const __attribute__((address_space(3))) _Float16* p0,
                                          int kpad, int kbase) {
  int lane = threadIdx.x & 31;
  const __attribute__((address_space(3))) _Float16* p =
      p0 + (lane & 15) * kpad + kbase + ((lane & 16) ? 16 : 0);
  v16h b;
#pragma unroll
  for (int i = 0; i < 16; ++i) b[i] = p[i];
  return b;
}

#define LDS_PTR(x) ((__attribute__((address_space(3))) _Float16*)(x))

// ---------------- node GVP (gv): s_tmp = relu([s0,sh]@Wm^T+b), V_tmp = |Vmu|*Vmu ----------------
__global__ __launch_bounds__(128) void node_gvp_kernel(
    const float* __restrict__ node_attrs,
    const float* __restrict__ Wh, const float* __restrict__ Wmu,
    const float* __restrict__ Wm_w, const float* __restrict__ Wm_b,
    float* __restrict__ s_tmp, float* __restrict__ V_tmp)
{
  __shared__ _Float16 wWh[32 * 32];     // gv_Wh (32x16) padded K->32
  __shared__ _Float16 wWmu[16 * 32];
  __shared__ _Float16 wWm[48 * 96];     // gv_Wm_w (48x80) padded K->96
  __shared__ float    bM[48];
  __shared__ _Float16 sS[WPB][16 * 96];
  __shared__ _Float16 sV[WPB][3][16 * 32];
  __shared__ _Float16 sVh[WPB][16 * 32];

  int tid = threadIdx.x;
  for (int i = tid; i < 32 * 32; i += 128) {
    int k = i & 31;
    wWh[i] = (k < 16) ? (_Float16)Wh[(i >> 5) * 16 + k] : (_Float16)0.f;
  }
  for (int i = tid; i < 16 * 32; i += 128) wWmu[i] = (_Float16)Wmu[i];
  for (int i = tid; i < 48 * 96; i += 128) {
    int n = i / 96, k = i - n * 96;
    wWm[i] = (k < 80) ? (_Float16)Wm_w[n * 80 + k] : (_Float16)0.f;
  }
  for (int i = tid; i < 48; i += 128) bM[i] = Wm_b[i];
  __syncthreads();

  int lane = tid & 31, wave = tid >> 5;
  int mrow = (lane & 16) ? 8 : 0;
  int ncol = lane & 15;

  // constant LDS regions: scat pad cols 80..95, vcat cols 16..31 (once per wave)
  for (int i = lane; i < 16 * 16; i += 32) sS[wave][(i >> 4) * 96 + 80 + (i & 15)] = (_Float16)0.f;
  for (int i = lane; i < 3 * 16 * 16; i += 32) {
    int d = i >> 8, r = (i >> 4) & 15, c = i & 15;
    sV[wave][d][r * 32 + 16 + c] = (_Float16)0.f;
  }

  // hoist loop-invariant B operands
  v16h bWh[2], bMu;
#pragma unroll
  for (int nt = 0; nt < 2; ++nt) bWh[nt] = lds_tileB(LDS_PTR(wWh) + nt * 16 * 32, 32, 0);
  bMu = lds_tileB(LDS_PTR(wWmu), 32, 0);

  int ntiles = N_NODES / 16;
  int wgid = blockIdx.x * WPB + wave;
  int nw = gridDim.x * WPB;
  for (int t = wgid; t < ntiles; t += nw) {
    int i0 = t * 16;
    {
      int h = lane >> 4, row = lane & 15;
      int cb = h * 8;
      const float4* na4 = (const float4*)(node_attrs + (long)(i0 + row) * NODE_DIM);
#pragma unroll
      for (int q = 0; q < 6; ++q) {                 // scalars 0..47
        float4 v = na4[h * 6 + q];
        int k = h * 24 + q * 4;
        sS[wave][row * 96 + k + 0] = (_Float16)v.x;
        sS[wave][row * 96 + k + 1] = (_Float16)v.y;
        sS[wave][row * 96 + k + 2] = (_Float16)v.z;
        sS[wave][row * 96 + k + 3] = (_Float16)v.w;
      }
#pragma unroll
      for (int q = 0; q < 6; ++q) {                 // vectors 48..95 -> (c,d) scatter
        float4 v = na4[12 + h * 6 + q];
#pragma unroll
        for (int j = 0; j < 4; ++j) {
          const int f = q * 4 + j;                  // compile-time after unroll
          sV[wave][f % 3][row * 32 + cb + f / 3] = (_Float16)f4_elem(v, j);
        }
      }
    }
    __builtin_amdgcn_wave_barrier();

    v8f vh[3][2];
    v16h aV[3];
#pragma unroll
    for (int d = 0; d < 3; ++d) aV[d] = lds_tileA(LDS_PTR(sV[wave][d]), 32, 0);
#pragma unroll
    for (int d = 0; d < 3; ++d)
#pragma unroll
      for (int nt = 0; nt < 2; ++nt) {
        v8f z = {};
        vh[d][nt] = wmma_f16(aV[d], bWh[nt], z);
      }
#pragma unroll
    for (int nt = 0; nt < 2; ++nt)
#pragma unroll
      for (int r = 0; r < 8; ++r) {
        float q = vh[0][nt][r]*vh[0][nt][r] + vh[1][nt][r]*vh[1][nt][r] + vh[2][nt][r]*vh[2][nt][r];
        sS[wave][(mrow + r) * 96 + 48 + nt * 16 + ncol] = (_Float16)fsqrt(q);
      }
    v8f vmu[3];
#pragma unroll
    for (int d = 0; d < 3; ++d) {
#pragma unroll
      for (int nt = 0; nt < 2; ++nt)
#pragma unroll
        for (int r = 0; r < 8; ++r)
          sVh[wave][(mrow + r) * 32 + nt * 16 + ncol] = (_Float16)vh[d][nt][r];
      __builtin_amdgcn_wave_barrier();
      v16h a = lds_tileA(LDS_PTR(sVh[wave]), 32, 0);
      v8f z = {};
      vmu[d] = wmma_f16(a, bMu, z);
    }
#pragma unroll
    for (int r = 0; r < 8; ++r) {
      int i = i0 + mrow + r;
      float x0 = vmu[0][r], x1 = vmu[1][r], x2 = vmu[2][r];
      float nm = fsqrt(x0*x0 + x1*x1 + x2*x2);
      float* vt = V_tmp + (long)i * 48 + ncol * 3;
      vt[0] = nm * x0; vt[1] = nm * x1; vt[2] = nm * x2;
    }
    __builtin_amdgcn_wave_barrier();
#pragma unroll
    for (int nt = 0; nt < 3; ++nt) {
      v8f acc = {};
#pragma unroll
      for (int kt = 0; kt < 3; ++kt) {
        v16h a = lds_tileA(LDS_PTR(sS[wave]), 96, kt * 32);
        v16h b = lds_tileB(LDS_PTR(wWm) + nt * 16 * 96, 96, kt * 32);
        acc = wmma_f16(a, b, acc);
      }
      int n = nt * 16 + ncol;
      float bias = bM[n];
#pragma unroll
      for (int r = 0; r < 8; ++r) {
        int i = i0 + mrow + r;
        float v = acc[r] + bias;
        s_tmp[(long)i * 48 + n] = v > 0.f ? v : 0.f;
      }
    }
  }
}

// ---------------- edge kernel: gather, gve message GVP (-> atomic segment sum), ge edge GVP ----------------
__global__ __launch_bounds__(128) void edge_kernel(
    const float* __restrict__ node2,
    const float* __restrict__ edge_attrs,
    const int* __restrict__ edge_idx,
    const float* __restrict__ gve_Wh, const float* __restrict__ gve_Wmu,
    const float* __restrict__ gve_Wm_w, const float* __restrict__ gve_Wm_b,
    const float* __restrict__ ge_Wh, const float* __restrict__ ge_Wmu,
    const float* __restrict__ ge_Wm_w, const float* __restrict__ ge_Wm_b,
    float* __restrict__ upd, float* __restrict__ edge_out)
{
  __shared__ _Float16 wWh1[32*32], wWmu1[16*32], wWm1[48*128];
  __shared__ _Float16 wWh2[32*32], wWmu2[16*32], wWm2[32*128];
  __shared__ float b1[48], b2[32];
  __shared__ _Float16 sS[WPB][16*128];
  __shared__ _Float16 sV[WPB][3][16*32];
  __shared__ _Float16 sVh[WPB][16*32];

  int tid = threadIdx.x;
  for (int i = tid; i < 32*32; i += 128) { wWh1[i] = (_Float16)gve_Wh[i]; wWh2[i] = (_Float16)ge_Wh[i]; }
  for (int i = tid; i < 16*32; i += 128) { wWmu1[i] = (_Float16)gve_Wmu[i]; wWmu2[i] = (_Float16)ge_Wmu[i]; }
  for (int i = tid; i < 48*128; i += 128) {
    int n = i >> 7, k = i & 127;
    wWm1[i] = (k < 112) ? (_Float16)gve_Wm_w[n*112 + k] : (_Float16)0.f;
  }
  for (int i = tid; i < 32*128; i += 128) {
    int n = i >> 7, k = i & 127;
    wWm2[i] = (k < 112) ? (_Float16)ge_Wm_w[n*112 + k] : (_Float16)0.f;
  }
  for (int i = tid; i < 48; i += 128) b1[i] = gve_Wm_b[i];
  for (int i = tid; i < 32; i += 128) b2[i] = ge_Wm_b[i];
  __syncthreads();

  int lane = tid & 31, wave = tid >> 5;
  int mrow = (lane & 16) ? 8 : 0;
  int ncol = lane & 15;

  // constant scat pad cols 112..127 (once per wave)
  for (int i = lane; i < 16 * 16; i += 32) sS[wave][(i >> 4) * 128 + 112 + (i & 15)] = (_Float16)0.f;

  // hoist loop-invariant B operands (Wh / Wmu of both GVPs)
  v16h bWhA[2], bWhB[2], bMuA, bMuB;
#pragma unroll
  for (int nt = 0; nt < 2; ++nt) {
    bWhA[nt] = lds_tileB(LDS_PTR(wWh1) + nt * 16 * 32, 32, 0);
    bWhB[nt] = lds_tileB(LDS_PTR(wWh2) + nt * 16 * 32, 32, 0);
  }
  bMuA = lds_tileB(LDS_PTR(wWmu1), 32, 0);
  bMuB = lds_tileB(LDS_PTR(wWmu2), 32, 0);

  int ntiles = N_EDGES / 16;
  int wgid = blockIdx.x * WPB + wave;
  int nw = gridDim.x * WPB;
  for (int t = wgid; t < ntiles; t += nw) {
    int e0 = t * 16;

    // prefetch next tile's edge rows a trip ahead (speculative, 1B/lane covers 160B each)
    if (t + nw < ntiles) {
      const float* nxt = edge_attrs + (long)(t + nw) * 16 * EDGE_DIM;
      __builtin_prefetch(nxt + lane * 40, 0, 1);
    }

    {
      int h = lane >> 4, row = lane & 15;
      int cb = h * 8;
      int e = e0 + row;
      int src = edge_idx[N_EDGES + e];                 // edge_idx row 1: gather source
      const float4* nn4 = (const float4*)(node2 + (long)src * NODE_DIM);
      const float4* ea4 = (const float4*)(edge_attrs + (long)e * EDGE_DIM);
#pragma unroll
      for (int q = 0; q < 6; ++q) {                    // node scalars -> cols 0..47
        float4 v = nn4[h * 6 + q];
        int k = h * 24 + q * 4;
        sS[wave][row*128 + k + 0] = (_Float16)v.x;
        sS[wave][row*128 + k + 1] = (_Float16)v.y;
        sS[wave][row*128 + k + 2] = (_Float16)v.z;
        sS[wave][row*128 + k + 3] = (_Float16)v.w;
      }
#pragma unroll
      for (int q = 0; q < 4; ++q) {                    // edge scalars -> cols 48..79
        float4 v = ea4[h * 4 + q];
        int k = 48 + h * 16 + q * 4;
        sS[wave][row*128 + k + 0] = (_Float16)v.x;
        sS[wave][row*128 + k + 1] = (_Float16)v.y;
        sS[wave][row*128 + k + 2] = (_Float16)v.z;
        sS[wave][row*128 + k + 3] = (_Float16)v.w;
      }
#pragma unroll
      for (int q = 0; q < 6; ++q) {                    // node vectors -> vcat channels 0..15
        float4 v = nn4[12 + h * 6 + q];
#pragma unroll
        for (int j = 0; j < 4; ++j) {
          const int f = q * 4 + j;                     // compile-time after unroll
          sV[wave][f % 3][row * 32 + cb + f / 3] = (_Float16)f4_elem(v, j);
        }
      }
#pragma unroll
      for (int q = 0; q < 6; ++q) {                    // edge vectors -> vcat channels 16..31
        float4 v = ea4[8 + h * 6 + q];
#pragma unroll
        for (int j = 0; j < 4; ++j) {
          const int f = q * 4 + j;
          sV[wave][f % 3][row * 32 + 16 + cb + f / 3] = (_Float16)f4_elem(v, j);
        }
      }
    }
    __builtin_amdgcn_wave_barrier();

    int dst[8];
#pragma unroll
    for (int r = 0; r < 8; ++r) dst[r] = edge_idx[e0 + mrow + r];   // row 0: segment ids

    // A tiles shared between gve and ge
    v16h aV[3];
#pragma unroll
    for (int d = 0; d < 3; ++d) aV[d] = lds_tileA(LDS_PTR(sV[wave][d]), 32, 0);
    v16h aS0 = lds_tileA(LDS_PTR(sS[wave]), 128, 0);     // cols 0..31 (invariant vs sh)
    v16h aS1 = lds_tileA(LDS_PTR(sS[wave]), 128, 32);    // cols 32..63 (invariant vs sh)

    // ---- message GVP (gve) ----
    {
      v8f vh[3][2];
#pragma unroll
      for (int d = 0; d < 3; ++d)
#pragma unroll
        for (int nt = 0; nt < 2; ++nt) {
          v8f z = {};
          vh[d][nt] = wmma_f16(aV[d], bWhA[nt], z);
        }
#pragma unroll
      for (int nt = 0; nt < 2; ++nt)
#pragma unroll
        for (int r = 0; r < 8; ++r) {
          float q = vh[0][nt][r]*vh[0][nt][r] + vh[1][nt][r]*vh[1][nt][r] + vh[2][nt][r]*vh[2][nt][r];
          sS[wave][(mrow + r)*128 + 80 + nt*16 + ncol] = (_Float16)fsqrt(q);
        }
      v8f vmu[3];
#pragma unroll
      for (int d = 0; d < 3; ++d) {
#pragma unroll
        for (int nt = 0; nt < 2; ++nt)
#pragma unroll
          for (int r = 0; r < 8; ++r)
            sVh[wave][(mrow + r)*32 + nt*16 + ncol] = (_Float16)vh[d][nt][r];
        __builtin_amdgcn_wave_barrier();
        v16h a = lds_tileA(LDS_PTR(sVh[wave]), 32, 0);
        v8f z = {};
        vmu[d] = wmma_f16(a, bMuA, z);
      }
#pragma unroll
      for (int r = 0; r < 8; ++r) {
        float x0 = vmu[0][r], x1 = vmu[1][r], x2 = vmu[2][r];
        float nm = fsqrt(x0*x0 + x1*x1 + x2*x2);
        float* ub = upd + (long)dst[r]*NODE_DIM + 48 + ncol*3;
        atomicAdd(ub + 0, nm * x0);
        atomicAdd(ub + 1, nm * x1);
        atomicAdd(ub + 2, nm * x2);
      }
      __builtin_amdgcn_wave_barrier();
#pragma unroll
      for (int nt = 0; nt < 3; ++nt) {
        v8f acc = {};
        acc = wmma_f16(aS0, lds_tileB(LDS_PTR(wWm1) + nt*16*128, 128, 0),  acc);
        acc = wmma_f16(aS1, lds_tileB(LDS_PTR(wWm1) + nt*16*128, 128, 32), acc);
#pragma unroll
        for (int kt = 2; kt < 4; ++kt) {
          v16h a = lds_tileA(LDS_PTR(sS[wave]), 128, kt*32);
          v16h b = lds_tileB(LDS_PTR(wWm1) + nt*16*128, 128, kt*32);
          acc = wmma_f16(a, b, acc);
        }
        int n = nt*16 + ncol;
        float bias = b1[n];
#pragma unroll
        for (int r = 0; r < 8; ++r) {
          float v = acc[r] + bias; v = v > 0.f ? v : 0.f;
          atomicAdd(upd + (long)dst[r]*NODE_DIM + n, v);
        }
      }
    }

    // ---- edge-output GVP (ge) ----
    {
      v8f vh[3][2];
#pragma unroll
      for (int d = 0; d < 3; ++d)
#pragma unroll
        for (int nt = 0; nt < 2; ++nt) {
          v8f z = {};
          vh[d][nt] = wmma_f16(aV[d], bWhB[nt], z);
        }
#pragma unroll
      for (int nt = 0; nt < 2; ++nt)
#pragma unroll
        for (int r = 0; r < 8; ++r) {
          float q = vh[0][nt][r]*vh[0][nt][r] + vh[1][nt][r]*vh[1][nt][r] + vh[2][nt][r]*vh[2][nt][r];
          sS[wave][(mrow + r)*128 + 80 + nt*16 + ncol] = (_Float16)fsqrt(q);
        }
      v8f vmu[3];
#pragma unroll
      for (int d = 0; d < 3; ++d) {
#pragma unroll
        for (int nt = 0; nt < 2; ++nt)
#pragma unroll
          for (int r = 0; r < 8; ++r)
            sVh[wave][(mrow + r)*32 + nt*16 + ncol] = (_Float16)vh[d][nt][r];
        __builtin_amdgcn_wave_barrier();
        v16h a = lds_tileA(LDS_PTR(sVh[wave]), 32, 0);
        v8f z = {};
        vmu[d] = wmma_f16(a, bMuB, z);
      }
#pragma unroll
      for (int r = 0; r < 8; ++r) {
        int e = e0 + mrow + r;
        float x0 = vmu[0][r], x1 = vmu[1][r], x2 = vmu[2][r];
        float nm = fsqrt(x0*x0 + x1*x1 + x2*x2);
        float* eo = edge_out + (long)e*EDGE_DIM + 32 + ncol*3;
        eo[0] = nm*x0; eo[1] = nm*x1; eo[2] = nm*x2;
      }
      __builtin_amdgcn_wave_barrier();
#pragma unroll
      for (int nt = 0; nt < 2; ++nt) {
        v8f acc = {};
        acc = wmma_f16(aS0, lds_tileB(LDS_PTR(wWm2) + nt*16*128, 128, 0),  acc);
        acc = wmma_f16(aS1, lds_tileB(LDS_PTR(wWm2) + nt*16*128, 128, 32), acc);
#pragma unroll
        for (int kt = 2; kt < 4; ++kt) {
          v16h a = lds_tileA(LDS_PTR(sS[wave]), 128, kt*32);
          v16h b = lds_tileB(LDS_PTR(wWm2) + nt*16*128, 128, kt*32);
          acc = wmma_f16(a, b, acc);
        }
        int n = nt*16 + ncol;
        float bias = b2[n];
#pragma unroll
        for (int r = 0; r < 8; ++r) {
          int e = e0 + mrow + r;
          float v = acc[r] + bias; v = v > 0.f ? v : 0.f;
          edge_out[(long)e*EDGE_DIM + n] = v;
        }
      }
    }
  }
}

// ---------------- elementwise passes ----------------
__global__ void k_zero(float* __restrict__ p, long n) {
  long i = (long)blockIdx.x * blockDim.x + threadIdx.x;
  if (i < n) p[i] = 0.f;
}

__global__ void k_post1(const float* __restrict__ node_attrs, const float* __restrict__ s_tmp,
                        const float* __restrict__ V_tmp, const float* __restrict__ ln_g,
                        const float* __restrict__ ln_b, float* __restrict__ node2,
                        float* __restrict__ sums)
{
  int j = blockIdx.x * blockDim.x + threadIdx.x;
  if (j >= N_NODES) return;
  const float* na = node_attrs + (long)j * NODE_DIM;
  float x[48]; float m = 0.f;
  for (int n = 0; n < 48; ++n) { x[n] = s_tmp[(long)j*48 + n] + na[n]; m += x[n]; }
  m *= (1.f/48.f);
  float v = 0.f;
  for (int n = 0; n < 48; ++n) { float d = x[n]-m; v += d*d; }
  v *= (1.f/48.f);
  float inv = frsqrt(v + 1e-5f);
  float* o = node2 + (long)j * NODE_DIM;
  for (int n = 0; n < 48; ++n) o[n] = (x[n]-m)*inv*ln_g[n] + ln_b[n];
  float ss = 0.f;
  for (int k = 0; k < 48; ++k) { float w = V_tmp[(long)j*48 + k] + na[48 + k]; o[48 + k] = w; ss += w*w; }
  atomicAdd(&sums[0], ss);
}

__global__ void k_scale1(float* __restrict__ node2, const float* __restrict__ sums) {
  long i = (long)blockIdx.x * blockDim.x + threadIdx.x;
  if (i >= (long)N_NODES * 48) return;
  float sc = frsqrt(fsqrt(sums[0]) * (1.f/16.f));
  long j = i / 48, k = i - j*48;
  node2[j*NODE_DIM + 48 + k] *= sc;
}

__global__ void k_post2(const float* __restrict__ node2, const float* __restrict__ upd,
                        const float* __restrict__ ln_g, const float* __restrict__ ln_b,
                        float* __restrict__ out_node, float* __restrict__ VuBuf,
                        float* __restrict__ sums)
{
  int j = blockIdx.x * blockDim.x + threadIdx.x;
  if (j >= N_NODES) return;
  const float* n2 = node2 + (long)j * NODE_DIM;
  const float* u  = upd + (long)j * NODE_DIM;
  float x[48]; float m = 0.f;
  for (int n = 0; n < 48; ++n) { x[n] = u[n]*(1.f/30.f) + n2[n]; m += x[n]; }
  m *= (1.f/48.f);
  float v = 0.f;
  for (int n = 0; n < 48; ++n) { float d = x[n]-m; v += d*d; }
  v *= (1.f/48.f);
  float inv = frsqrt(v + 1e-5f);
  float* o = out_node + (long)j * NODE_DIM;
  for (int n = 0; n < 48; ++n) {
    float su = (x[n]-m)*inv*ln_g[n] + ln_b[n];
    o[n] = n2[n] + su;
  }
  float ss = 0.f;
  for (int k = 0; k < 48; ++k) {
    float w = u[48 + k]*(1.f/30.f) + n2[48 + k];
    VuBuf[(long)j*48 + k] = w; ss += w*w;
  }
  atomicAdd(&sums[1], ss);
}

__global__ void k_scale2(const float* __restrict__ node2, const float* __restrict__ VuBuf,
                         const float* __restrict__ sums, float* __restrict__ out_node) {
  long i = (long)blockIdx.x * blockDim.x + threadIdx.x;
  if (i >= (long)N_NODES * 48) return;
  float sc = frsqrt(fsqrt(sums[1]) * (1.f/16.f));
  long j = i / 48, k = i - j*48;
  out_node[j*NODE_DIM + 48 + k] = node2[j*NODE_DIM + 48 + k] + VuBuf[i]*sc;
}

extern "C" void kernel_launch(void* const* d_in, const int* in_sizes, int n_in,
                              void* d_out, int out_size, void* d_ws, size_t ws_size,
                              hipStream_t stream)
{
  const float* node_attrs = (const float*)d_in[0];
  const float* edge_attrs = (const float*)d_in[1];
  const int*   edge_idx   = (const int*)d_in[2];
  const float* gv_Wh    = (const float*)d_in[3];
  const float* gv_Wmu   = (const float*)d_in[4];
  const float* gv_Wm_w  = (const float*)d_in[5];
  const float* gv_Wm_b  = (const float*)d_in[6];
  const float* gve_Wh   = (const float*)d_in[7];
  const float* gve_Wmu  = (const float*)d_in[8];
  const float* gve_Wm_w = (const float*)d_in[9];
  const float* gve_Wm_b = (const float*)d_in[10];
  const float* ge_Wh    = (const float*)d_in[11];
  const float* ge_Wmu   = (const float*)d_in[12];
  const float* ge_Wm_w  = (const float*)d_in[13];
  const float* ge_Wm_b  = (const float*)d_in[14];
  const float* ln_g     = (const float*)d_in[15];
  const float* ln_b     = (const float*)d_in[16];

  float* ws = (float*)d_ws;
  const long offNode2 = 0;
  const long offUpd   = (long)N_NODES * NODE_DIM;
  const long offSums  = offUpd + (long)N_NODES * NODE_DIM;   // contiguous with upd
  const long offStmp  = offSums + 16;
  const long offVtmp  = offStmp + (long)N_NODES * 48;
  const long offVu    = offVtmp + (long)N_NODES * 48;
  float* node2 = ws + offNode2;
  float* upd   = ws + offUpd;
  float* sums  = ws + offSums;
  float* s_tmp = ws + offStmp;
  float* V_tmp = ws + offVtmp;
  float* VuBuf = ws + offVu;

  float* out_node = (float*)d_out;
  float* out_edge = out_node + (long)N_NODES * NODE_DIM;

  long zcount = (long)N_NODES * NODE_DIM + 16;  // upd + sums
  k_zero<<<(int)((zcount + 255) / 256), 256, 0, stream>>>(upd, zcount);

  node_gvp_kernel<<<256, 128, 0, stream>>>(node_attrs, gv_Wh, gv_Wmu, gv_Wm_w, gv_Wm_b,
                                           s_tmp, V_tmp);
  k_post1<<<(N_NODES + 255) / 256, 256, 0, stream>>>(node_attrs, s_tmp, V_tmp, ln_g, ln_b,
                                                     node2, sums);
  k_scale1<<<(int)(((long)N_NODES * 48 + 255) / 256), 256, 0, stream>>>(node2, sums);

  edge_kernel<<<1024, 128, 0, stream>>>(node2, edge_attrs, edge_idx,
                                        gve_Wh, gve_Wmu, gve_Wm_w, gve_Wm_b,
                                        ge_Wh, ge_Wmu, ge_Wm_w, ge_Wm_b,
                                        upd, out_edge);

  k_post2<<<(N_NODES + 255) / 256, 256, 0, stream>>>(node2, upd, ln_g, ln_b,
                                                     out_node, VuBuf, sums);
  k_scale2<<<(int)(((long)N_NODES * 48 + 255) / 256), 256, 0, stream>>>(node2, VuBuf, sums,
                                                                        out_node);
}
